// YuKeMLPFLOW_onlyX_seperateZ_37460704756278
// MI455X (gfx1250) — matrix operand order repaired and checked by reading, not compiled, not run on hardware
//
#include <hip/hip_runtime.h>
#include <math.h>

#define LATENT 12
#define HID    64
#define SLOPE  0.2f

typedef _Float16 v16h __attribute__((ext_vector_type(16)));
typedef _Float16 h8   __attribute__((ext_vector_type(8)));
typedef _Float16 hf2  __attribute__((ext_vector_type(2)));
typedef short    s2   __attribute__((ext_vector_type(2)));
typedef float    v8f  __attribute__((ext_vector_type(8)));

__device__ __forceinline__ v8f wmma_f16(v16h a, v16h b, v8f c) {
    // D = A(16x32 f16) * B(32x16 f16) + C(16x16 f32)
    return __builtin_amdgcn_wmma_f32_16x16x32_f16(
        /*neg_a=*/false, a, /*neg_b=*/false, b,
        /*c_mod=*/(short)0, c, /*reuse_a=*/false, /*reuse_b=*/false);
}

__device__ __forceinline__ v16h cat8(h8 lo, h8 hi) {
    return __builtin_shufflevector(lo, hi, 0,1,2,3,4,5,6,7,8,9,10,11,12,13,14,15);
}

// packed leaky-relu: hv = max(pre, S*pre)  -> v_pk_mul_f16 + v_pk_max_num_f16
__device__ __forceinline__ hf2 lrelu_pk(hf2 pre) {
    const hf2 s = {(_Float16)SLOPE, (_Float16)SLOPE};
    return __builtin_elementwise_max(pre, pre * s);
}
// packed derivative gate: d = pre>0 ? 1.0h : 0.2h, via per-half sign mask + bfi
__device__ __forceinline__ hf2 dgate_pk(hf2 pre) {
    s2 m = __builtin_bit_cast(s2, pre) >> 15;          // v_pk_ashrrev_i16
    const s2 sb = {(short)0x3266, (short)0x3266};       // f16(0.2)
    const s2 ob = {(short)0x3C00, (short)0x3C00};       // f16(1.0)
    s2 d = (m & sb) | (~m & ob);                        // v_bitop3/bfi
    return __builtin_bit_cast(hf2, d);
}

__global__ void zero_logdet_kernel(float* __restrict__ out, int B) {
    int i = blockIdx.x * blockDim.x + threadIdx.x;
    if (i < B) out[(size_t)B * LATENT + i] = 0.0f;
}

__launch_bounds__(256)
__global__ void flow_wmma_kernel(const float* __restrict__ x,
                                 const float* __restrict__ W0,
                                 const float* __restrict__ b0,
                                 const float* __restrict__ W1,
                                 const float* __restrict__ b1,
                                 const float* __restrict__ W2,
                                 const float* __restrict__ b2,
                                 const float* __restrict__ W3,
                                 const float* __restrict__ b3,
                                 float* __restrict__ out,
                                 int B) {
    // f16 weights in [g][h] row-major -> B-fragment loads are contiguous
    __shared__ __align__(16) _Float16 w1h[HID * HID];
    __shared__ __align__(16) _Float16 w2h[HID * HID];
    // per-wave activation scratch for D->A relayout (h and tangent)
    __shared__ __align__(16) _Float16 actH[8][16 * HID];
    __shared__ __align__(16) _Float16 actT[8][16 * HID];
    __shared__ float b1s[HID], b2s[HID];
    __shared__ __align__(16) _Float16 w0h[HID];  // f16 W0 (layer0 + tangent)
    __shared__ __align__(16) _Float16 b0h[HID];  // f16 b0
    __shared__ __align__(16) _Float16 w3h[HID];  // f16 W3 for WMMA output dot

    const int l   = blockIdx.y;       // latent index
    const int tid = threadIdx.x;

    // ---- cooperative weight staging (f32 -> f16 for GEMM weights) ----
    {
        const float* w1p = W1 + (size_t)l * HID * HID;
        const float* w2p = W2 + (size_t)l * HID * HID;
        for (int i = tid; i < HID * HID; i += 256) {
            w1h[i] = (_Float16)w1p[i];
            w2h[i] = (_Float16)w2p[i];
        }
        for (int i = tid; i < HID; i += 256) {
            w0h[i] = (_Float16)W0[(size_t)l * HID + i];
            b0h[i] = (_Float16)b0[(size_t)l * HID + i];
            b1s[i] = b1[(size_t)l * HID + i];
            b2s[i] = b2[(size_t)l * HID + i];
            w3h[i] = (_Float16)W3[(size_t)l * HID + i];
        }
    }
    __syncthreads();

    const int wave = tid >> 5;
    const int lane = tid & 31;
    const int n    = lane & 15;   // column index for B/C/D, row index for A
    const int hlf  = lane >> 4;   // lane-group select (K-half / M-half)
    const int base = blockIdx.x * 128 + wave * 16;

    _Float16* aH = actH[wave];
    _Float16* aT = actT[wave];

    // ---- layer 0: 1 -> 64, packed f16, built directly in A-frag layout ----
    // A 16x32 f16: lane<16 holds K=k0..k0+7 (frag[0..7]) and k0+16..k0+23
    // (frag[8..15]); lane>=16 holds K=k0+8.. and k0+24..
    const float xv = x[(size_t)(base + n) * LATENT + l];
    const _Float16 xh = (_Float16)xv;
    const hf2 xh2 = {xh, xh};
    v16h a0h[2], a0t[2];
#pragma unroll
    for (int kb = 0; kb < 2; ++kb) {
#pragma unroll
        for (int jj = 0; jj < 8; ++jj) {          // pairs of frag elements
            const int j0 = jj * 2;
            const int h0 = kb * 32 + (j0 < 8 ? hlf * 8 + j0
                                             : 16 + hlf * 8 + (j0 - 8));
            const hf2 w0p = *(const hf2*)&w0h[h0];
            const hf2 b0p = *(const hf2*)&b0h[h0];
            hf2 pre2 = xh2 * w0p + b0p;            // v_pk_fma_f16 (contract)
            hf2 hv2  = lrelu_pk(pre2);
            hf2 tv2  = dgate_pk(pre2) * w0p;       // tangent of x is 1
            a0h[kb][j0]     = hv2[0];
            a0h[kb][j0 + 1] = hv2[1];
            a0t[kb][j0]     = tv2[0];
            a0t[kb][j0 + 1] = tv2[1];
        }
    }

    // ---- layer 1: WMMA phase (all tiles), then activation phase ----
    v8f accH[4], accT[4];
#pragma unroll
    for (int gb = 0; gb < 4; ++gb) {
        const int g = gb * 16 + n;
        v16h bf[2];
#pragma unroll
        for (int kb = 0; kb < 2; ++kb) {
            const h8* p = (const h8*)&w1h[g * HID + kb * 32 + hlf * 8];
            bf[kb] = cat8(p[0], p[2]);
        }
        const float bb = b1s[g];
#pragma unroll
        for (int r = 0; r < 8; ++r) { accH[gb][r] = bb; accT[gb][r] = 0.f; }
#pragma unroll
        for (int kb = 0; kb < 2; ++kb) {
            accH[gb] = wmma_f16(a0h[kb], bf[kb], accH[gb]);
            accT[gb] = wmma_f16(a0t[kb], bf[kb], accT[gb]);
        }
    }
#pragma unroll
    for (int gb = 0; gb < 4; ++gb) {
        const int g = gb * 16 + n;
#pragma unroll
        for (int rr = 0; rr < 4; ++rr) {         // pairs of D rows
            const int r0 = rr * 2;
            hf2 pre2 = {(_Float16)accH[gb][r0], (_Float16)accH[gb][r0 + 1]};
            hf2 tin2 = {(_Float16)accT[gb][r0], (_Float16)accT[gb][r0 + 1]};
            hf2 hv2  = lrelu_pk(pre2);
            hf2 tv2  = dgate_pk(pre2) * tin2;
            const int m0 = r0 + 8 * hlf;
            aH[m0 * HID + g]       = hv2[0];
            aH[(m0 + 1) * HID + g] = hv2[1];
            aT[m0 * HID + g]       = tv2[0];
            aT[(m0 + 1) * HID + g] = tv2[1];
        }
    }

    // ---- reload layer-1 activations as A fragments (wave-private LDS) ----
    v16h a1h[2], a1t[2];
#pragma unroll
    for (int kb = 0; kb < 2; ++kb) {
        const int o0 = n * HID + kb * 32 + hlf * 8;
        a1h[kb] = cat8(*(const h8*)&aH[o0], *(const h8*)&aH[o0 + 16]);
        a1t[kb] = cat8(*(const h8*)&aT[o0], *(const h8*)&aT[o0 + 16]);
    }

    // ---- layer 2: WMMA phase, then activation -> LDS (reuse scratch) ----
#pragma unroll
    for (int gb = 0; gb < 4; ++gb) {
        const int g = gb * 16 + n;
        v16h bf[2];
#pragma unroll
        for (int kb = 0; kb < 2; ++kb) {
            const h8* p = (const h8*)&w2h[g * HID + kb * 32 + hlf * 8];
            bf[kb] = cat8(p[0], p[2]);
        }
        const float bb = b2s[g];
#pragma unroll
        for (int r = 0; r < 8; ++r) { accH[gb][r] = bb; accT[gb][r] = 0.f; }
#pragma unroll
        for (int kb = 0; kb < 2; ++kb) {
            accH[gb] = wmma_f16(a1h[kb], bf[kb], accH[gb]);
            accT[gb] = wmma_f16(a1t[kb], bf[kb], accT[gb]);
        }
    }
#pragma unroll
    for (int gb = 0; gb < 4; ++gb) {
        const int g = gb * 16 + n;
#pragma unroll
        for (int rr = 0; rr < 4; ++rr) {
            const int r0 = rr * 2;
            hf2 pre2 = {(_Float16)accH[gb][r0], (_Float16)accH[gb][r0 + 1]};
            hf2 tin2 = {(_Float16)accT[gb][r0], (_Float16)accT[gb][r0 + 1]};
            hf2 hv2  = lrelu_pk(pre2);
            hf2 tv2  = dgate_pk(pre2) * tin2;
            const int m0 = r0 + 8 * hlf;
            aH[m0 * HID + g]       = hv2[0];
            aH[(m0 + 1) * HID + g] = hv2[1];
            aT[m0 * HID + g]       = tv2[0];
            aT[(m0 + 1) * HID + g] = tv2[1];
        }
    }

    // ---- output dot (64 -> 1) as WMMA with W3 replicated across columns ----
    // B[k][n] = W3[k] for all n  =>  D[m][n] = sum_k h2[m][k]*W3[k] in every
    // lane: no cross-lane reduction needed.
    v16h a2h[2], a2t[2], bw3[2];
#pragma unroll
    for (int kb = 0; kb < 2; ++kb) {
        const int o0 = n * HID + kb * 32 + hlf * 8;
        a2h[kb] = cat8(*(const h8*)&aH[o0], *(const h8*)&aH[o0 + 16]);
        a2t[kb] = cat8(*(const h8*)&aT[o0], *(const h8*)&aT[o0 + 16]);
        const h8* pw = (const h8*)&w3h[kb * 32 + hlf * 8];  // lane-broadcast
        bw3[kb] = cat8(pw[0], pw[2]);
    }
    v8f accR = {}, accJ = {};
#pragma unroll
    for (int kb = 0; kb < 2; ++kb) {
        accR = wmma_f16(a2h[kb], bw3[kb], accR);
        accJ = wmma_f16(a2t[kb], bw3[kb], accJ);
    }

    if (n == 0) {
        const float b3v = b3[l];
#pragma unroll
        for (int r = 0; r < 8; ++r) {
            const int b = base + 8 * hlf + r;
            out[(size_t)b * LATENT + l] = accR[r] + b3v;
            atomicAdd(&out[(size_t)B * LATENT + b], __logf(fabsf(accJ[r])));
        }
    }
}

extern "C" void kernel_launch(void* const* d_in, const int* in_sizes, int n_in,
                              void* d_out, int out_size, void* d_ws, size_t ws_size,
                              hipStream_t stream) {
    const float* x  = (const float*)d_in[0];
    const float* W0 = (const float*)d_in[1];
    const float* b0 = (const float*)d_in[2];
    const float* W1 = (const float*)d_in[3];
    const float* b1 = (const float*)d_in[4];
    const float* W2 = (const float*)d_in[5];
    const float* b2 = (const float*)d_in[6];
    const float* W3 = (const float*)d_in[7];
    const float* b3 = (const float*)d_in[8];
    float* out = (float*)d_out;

    const int B = in_sizes[0] / LATENT;   // 131072

    zero_logdet_kernel<<<(B + 255) / 256, 256, 0, stream>>>(out, B);

    dim3 grid(B / 128, LATENT);           // 8 waves * 16 rows per block
    flow_wmma_kernel<<<grid, 256, 0, stream>>>(x, W0, b0, W1, b1, W2, b2, W3, b3,
                                               out, B);
}
